// MLA_65326452572680
// MI455X (gfx1250) — matrix operand order repaired and checked by reading, compiled, not run on hardware
//
#include <hip/hip_runtime.h>
#include <hip/hip_bf16.h>

#define D_MODEL 2048
#define NUM_HEADS 16
#define NUM_KV 4
#define HEAD_DIM 128
#define LATENT 64
#define BB 4
#define TT 2048
#define MTOK (BB * TT)          // 8192 tokens
#define KV_N (2 * NUM_KV * HEAD_DIM)  // 1024 (k restored 512 + v restored 512)
#define LOG2E 1.44269504088896340736f

typedef __attribute__((ext_vector_type(16))) __bf16 v16bf;
typedef __attribute__((ext_vector_type(8)))  __bf16 v8bf;
typedef __attribute__((ext_vector_type(8)))  float  v8f;

__device__ __forceinline__ float fast_exp2(float x) {
#if __has_builtin(__builtin_amdgcn_exp2f)
  return __builtin_amdgcn_exp2f(x);
#else
  return exp2f(x);
#endif
}

// Build a 16-element bf16 fragment from two 16-byte chunks.
__device__ __forceinline__ v16bf ld16x2(const __bf16* p0, const __bf16* p1) {
  v8bf lo = *(const v8bf*)p0;
  v8bf hi = *(const v8bf*)p1;
  return __builtin_shufflevector(lo, hi, 0,1,2,3,4,5,6,7,8,9,10,11,12,13,14,15);
}

__device__ __forceinline__ v8f wmma_bf16(v16bf a, v16bf b, v8f c) {
  return __builtin_amdgcn_wmma_f32_16x16x32_bf16(false, a, false, b, (short)0, c,
                                                 false, false);
}

// ---------------------------------------------------------------- prep kernels

__global__ void cvt_bf16(const float* __restrict__ in, __bf16* __restrict__ out,
                         size_t n) {
  size_t i = (size_t)blockIdx.x * blockDim.x + threadIdx.x;
  if (i < n) out[i] = (__bf16)in[i];
}

// out[n][k] = in[k][n] * scale   (in: [K][N] row-major)
__global__ void transpose_cvt(const float* __restrict__ in, __bf16* __restrict__ out,
                              int K, int N, float scale) {
  size_t i = (size_t)blockIdx.x * blockDim.x + threadIdx.x;
  if (i >= (size_t)K * N) return;
  int n = (int)(i / K);
  int k = (int)(i % K);
  out[i] = (__bf16)(in[(size_t)k * N + n] * scale);
}

// eff[sel][i][j] = sum_l W2l[i][l] * Wfl[l][j]   (sel 0 = k, 1 = v)
__global__ void compose_eff(const float* __restrict__ Wk2l, const float* __restrict__ Wkfl,
                            const float* __restrict__ Wv2l, const float* __restrict__ Wvfl,
                            float* __restrict__ eff) {
  int i = blockIdx.x * blockDim.x + threadIdx.x;   // < 2*128*128
  if (i >= 2 * HEAD_DIM * HEAD_DIM) return;
  int sel = i >> 14;
  int r = (i >> 7) & 127;
  int c = i & 127;
  const float* A = sel ? Wv2l : Wk2l;   // [128][64]
  const float* B = sel ? Wvfl : Wkfl;   // [64][128]
  float acc = 0.f;
  for (int j = 0; j < LATENT; ++j) acc += A[r * LATENT + j] * B[j * HEAD_DIM + c];
  eff[i] = acc;
}

// Wkvt[n][k] (transposed, bf16). n in [0,512): k-path head h=n/128, out-dim j=n%128.
// n in [512,1024): v-path. Wkvt[n][k] = sum_i W[k][h*128+i] * eff[i][j]
__global__ void make_wkv_t(const float* __restrict__ Wk, const float* __restrict__ Wv,
                           const float* __restrict__ eff, __bf16* __restrict__ out) {
  size_t i = (size_t)blockIdx.x * blockDim.x + threadIdx.x;  // < 1024*2048
  if (i >= (size_t)KV_N * D_MODEL) return;
  int n = (int)(i >> 11);
  int k = (int)(i & 2047);
  int isv = n >> 9;
  int nn = n & 511;
  int hh = nn >> 7;
  int j = nn & 127;
  const float* W = isv ? Wv : Wk;                      // [2048][512]
  const float* E = eff + (size_t)isv * HEAD_DIM * HEAD_DIM;
  float acc = 0.f;
  for (int t = 0; t < HEAD_DIM; ++t)
    acc += W[(size_t)k * 512 + hh * 128 + t] * E[t * 128 + j];
  out[i] = (__bf16)acc;
}

// vt[((b*4+hkv)*128+d)][t] = kv[(b*T+t)][512 + hkv*128 + d]
__global__ void transpose_v(const __bf16* __restrict__ kvb, __bf16* __restrict__ vt) {
  size_t i = (size_t)blockIdx.x * blockDim.x + threadIdx.x;  // < 4*4*128*2048
  if (i >= (size_t)BB * NUM_KV * HEAD_DIM * TT) return;
  int t = (int)(i & 2047);
  size_t rest = i >> 11;
  int d = (int)(rest & 127);
  rest >>= 7;
  int hk = (int)(rest & 3);
  int b = (int)(rest >> 2);
  vt[i] = kvb[((size_t)(b * TT + t)) * KV_N + 512 + hk * 128 + d];
}

// ---------------------------------------------------------------- WMMA GEMM
// C[M,N] = A[M,K](bf16, row-major) * Bt[N,K](bf16, pre-transposed).
// 128 threads = 4 waves arranged 2x2; workgroup tile 128x128, wave tile 64x64.
template <typename OutT>
__global__ __launch_bounds__(128) void gemm_wmma(const __bf16* __restrict__ A,
                                                 const __bf16* __restrict__ Bt,
                                                 OutT* __restrict__ C,
                                                 int M, int N, int K) {
  const int lane = threadIdx.x & 31;
  const int wave = threadIdx.x >> 5;
  const int l15 = lane & 15;
  const int half = lane >> 4;
  const int bm = blockIdx.y * 128 + (wave >> 1) * 64;
  const int bn = blockIdx.x * 128 + (wave & 1) * 64;

  v8f acc[4][4] = {};

  for (int k0 = 0; k0 < K; k0 += 32) {
    v16bf a[4], b[4];
#pragma unroll
    for (int mi = 0; mi < 4; ++mi) {
      const __bf16* p = A + (size_t)(bm + mi * 16 + l15) * K + k0 + half * 16;
      a[mi] = ld16x2(p, p + 8);
    }
#pragma unroll
    for (int ni = 0; ni < 4; ++ni) {
      const __bf16* p = Bt + (size_t)(bn + ni * 16 + l15) * K + k0 + half * 16;
      b[ni] = ld16x2(p, p + 8);
    }
#pragma unroll
    for (int mi = 0; mi < 4; ++mi)
#pragma unroll
      for (int ni = 0; ni < 4; ++ni)
        acc[mi][ni] = wmma_bf16(a[mi], b[ni], acc[mi][ni]);
  }

#pragma unroll
  for (int mi = 0; mi < 4; ++mi)
#pragma unroll
    for (int ni = 0; ni < 4; ++ni)
#pragma unroll
      for (int r = 0; r < 8; ++r)
        C[(size_t)(bm + mi * 16 + r + half * 8) * N + bn + ni * 16 + l15] =
            (OutT)acc[mi][ni][r];
}

// ---------------------------------------------------------------- attention
// Flash attention, no mask. Scale 1/sqrt(128) pre-folded into Wq_t.
// grid: (T/64, B*H). 128 threads = 4 waves; each wave owns 16 query rows.
__global__ __launch_bounds__(128) void attn_wmma(const __bf16* __restrict__ q,
                                                 const __bf16* __restrict__ kv,
                                                 const __bf16* __restrict__ vt,
                                                 __bf16* __restrict__ ctx) {
  __shared__ __bf16 plds[4][16][32];  // per-wave P staging (C-layout -> A-layout)

  const int lane = threadIdx.x & 31;
  const int wave = threadIdx.x >> 5;
  const int l15 = lane & 15;
  const int half = lane >> 4;
  const int bh = blockIdx.y;
  const int b = bh >> 4;
  const int h = bh & 15;
  const int hkv = h >> 2;
  const int q0 = blockIdx.x * 64 + wave * 16;

  // resident Q fragments (16 rows x 128 d = 4 A-fragments)
  const __bf16* qbase = q + (size_t)(b * TT + q0 + l15) * D_MODEL + h * HEAD_DIM;
  v16bf qa[4];
#pragma unroll
  for (int ks = 0; ks < 4; ++ks) {
    const __bf16* p = qbase + ks * 32 + half * 16;
    qa[ks] = ld16x2(p, p + 8);
  }

  v8f o[8] = {};
  float mrun[8], lrun[8];
#pragma unroll
  for (int r = 0; r < 8; ++r) { mrun[r] = -3.0e38f; lrun[r] = 0.f; }

  const __bf16* kbase0 = kv + (size_t)(b * TT) * KV_N + hkv * HEAD_DIM;
  const __bf16* vbase0 = vt + (size_t)((b * NUM_KV + hkv) * HEAD_DIM) * TT;

  for (int kt = 0; kt < TT; kt += 32) {
    // S = Q * K^T for 32 keys, two 16x16 tiles
    v8f s0 = {}, s1 = {};
#pragma unroll
    for (int ks = 0; ks < 4; ++ks) {
      const __bf16* p0 = kbase0 + (size_t)(kt + l15) * KV_N + ks * 32 + half * 16;
      v16bf kb0 = ld16x2(p0, p0 + 8);
      s0 = wmma_bf16(qa[ks], kb0, s0);
      const __bf16* p1 = kbase0 + (size_t)(kt + 16 + l15) * KV_N + ks * 32 + half * 16;
      v16bf kb1 = ld16x2(p1, p1 + 8);
      s1 = wmma_bf16(qa[ks], kb1, s1);
    }

    // online softmax; rows live in 16-lane halves (m = r + half*8)
#pragma unroll
    for (int r = 0; r < 8; ++r) {
      float a0 = s0[r], a1 = s1[r];
      float t = fmaxf(a0, a1);
#pragma unroll
      for (int m = 1; m < 16; m <<= 1) t = fmaxf(t, __shfl_xor(t, m, 32));
      float newm = fmaxf(mrun[r], t);
      float scale = fast_exp2((mrun[r] - newm) * LOG2E);
      float p0 = fast_exp2((a0 - newm) * LOG2E);
      float p1 = fast_exp2((a1 - newm) * LOG2E);
      float rs = p0 + p1;
#pragma unroll
      for (int m = 1; m < 16; m <<= 1) rs += __shfl_xor(rs, m, 32);
      lrun[r] = lrun[r] * scale + rs;
      mrun[r] = newm;
#pragma unroll
      for (int ni = 0; ni < 8; ++ni) o[ni][r] *= scale;
      plds[wave][r + half * 8][l15] = (__bf16)p0;
      plds[wave][r + half * 8][16 + l15] = (__bf16)p1;
    }

    // re-read P in A-fragment layout (DS ops are in-order within a wave)
    const __bf16* pp = &plds[wave][l15][half * 16];
    v16bf pa = ld16x2(pp, pp + 8);

    // O += P * V  (V pre-transposed: row d contiguous in keys)
#pragma unroll
    for (int ni = 0; ni < 8; ++ni) {
      const __bf16* vp = vbase0 + (size_t)(ni * 16 + l15) * TT + kt + half * 16;
      v16bf vb = ld16x2(vp, vp + 8);
      o[ni] = wmma_bf16(pa, vb, o[ni]);
    }
  }

  float inv[8];
#pragma unroll
  for (int r = 0; r < 8; ++r) inv[r] = 1.f / lrun[r];

  __bf16* cbase = ctx + (size_t)(b * TT + q0) * D_MODEL + h * HEAD_DIM;
#pragma unroll
  for (int ni = 0; ni < 8; ++ni)
#pragma unroll
    for (int r = 0; r < 8; ++r)
      cbase[(size_t)(r + half * 8) * D_MODEL + ni * 16 + l15] =
          (__bf16)(o[ni][r] * inv[r]);
}

// ---------------------------------------------------------------- launcher

extern "C" void kernel_launch(void* const* d_in, const int* in_sizes, int n_in,
                              void* d_out, int out_size, void* d_ws, size_t ws_size,
                              hipStream_t stream) {
  const float* x    = (const float*)d_in[0];
  const float* Wq   = (const float*)d_in[1];
  const float* Wk   = (const float*)d_in[2];
  const float* Wv   = (const float*)d_in[3];
  const float* Wk2l = (const float*)d_in[4];
  const float* Wv2l = (const float*)d_in[5];
  const float* Wkfl = (const float*)d_in[6];
  const float* Wvfl = (const float*)d_in[7];
  const float* Wo   = (const float*)d_in[8];
  float* out = (float*)d_out;

  char* ws = (char*)d_ws;
  auto carve = [&](size_t bytes) -> char* {
    char* p = ws;
    ws += (bytes + 255) & ~(size_t)255;
    return p;
  };
  __bf16* x_bf   = (__bf16*)carve((size_t)MTOK * D_MODEL * 2);
  __bf16* q_bf   = (__bf16*)carve((size_t)MTOK * D_MODEL * 2);
  __bf16* kv_bf  = (__bf16*)carve((size_t)MTOK * KV_N * 2);
  __bf16* ctx_bf = (__bf16*)carve((size_t)MTOK * D_MODEL * 2);
  __bf16* vtb    = (__bf16*)carve((size_t)BB * NUM_KV * HEAD_DIM * TT * 2);
  __bf16* wq_t   = (__bf16*)carve((size_t)D_MODEL * D_MODEL * 2);
  __bf16* wo_t   = (__bf16*)carve((size_t)D_MODEL * D_MODEL * 2);
  __bf16* wkv_t  = (__bf16*)carve((size_t)KV_N * D_MODEL * 2);
  float*  eff    = (float*)carve((size_t)2 * HEAD_DIM * HEAD_DIM * 4);

  const float qscale = 0.08838834764831845f;  // 1/sqrt(128)

  {
    size_t n = (size_t)MTOK * D_MODEL;
    cvt_bf16<<<(unsigned)((n + 255) / 256), 256, 0, stream>>>(x, x_bf, n);
  }
  {
    size_t n = (size_t)D_MODEL * D_MODEL;
    transpose_cvt<<<(unsigned)((n + 255) / 256), 256, 0, stream>>>(Wq, wq_t, D_MODEL, D_MODEL, qscale);
    transpose_cvt<<<(unsigned)((n + 255) / 256), 256, 0, stream>>>(Wo, wo_t, D_MODEL, D_MODEL, 1.0f);
  }
  compose_eff<<<(2 * HEAD_DIM * HEAD_DIM + 255) / 256, 256, 0, stream>>>(Wk2l, Wkfl, Wv2l, Wvfl, eff);
  {
    size_t n = (size_t)KV_N * D_MODEL;
    make_wkv_t<<<(unsigned)((n + 255) / 256), 256, 0, stream>>>(Wk, Wv, eff, wkv_t);
  }

  gemm_wmma<__bf16><<<dim3(D_MODEL / 128, MTOK / 128), 128, 0, stream>>>(
      x_bf, wq_t, q_bf, MTOK, D_MODEL, D_MODEL);
  gemm_wmma<__bf16><<<dim3(KV_N / 128, MTOK / 128), 128, 0, stream>>>(
      x_bf, wkv_t, kv_bf, MTOK, KV_N, D_MODEL);

  {
    size_t n = (size_t)BB * NUM_KV * HEAD_DIM * TT;
    transpose_v<<<(unsigned)((n + 255) / 256), 256, 0, stream>>>(kv_bf, vtb);
  }

  attn_wmma<<<dim3(TT / 64, BB * NUM_HEADS), 128, 0, stream>>>(q_bf, kv_bf, vtb, ctx_bf);

  gemm_wmma<float><<<dim3(D_MODEL / 128, MTOK / 128), 128, 0, stream>>>(
      ctx_bf, wo_t, out, MTOK, D_MODEL, D_MODEL);
}
